// MSSA_71296457113965
// MI455X (gfx1250) — matrix operand order repaired and checked by reading, not compiled, not run on hardware
//
#include <hip/hip_runtime.h>
#include <hip/hip_bf16.h>
#include <math.h>

// ---------------------------------------------------------------------------
// MSSA on gfx1250: all GEMMs via v_wmma_f32_16x16x32_f16 (wave32).
// Pipeline:
//   wprep:  W[in][out] f32 -> Wt[out][in] f16   (x6)
//   gate :  s = gelu(X @ ws + bs)               (x2, wave-per-token reduction)
//   proj :  q/k row-major f16, v transposed f16 (x6, WMMA)
//   attn :  flash attention, 4 waves/WG sharing K/V tiles via LDS,
//           online softmax, gated residual epilogue (x2, WMMA)
// ---------------------------------------------------------------------------

typedef __attribute__((ext_vector_type(16))) _Float16 v16h;
typedef __attribute__((ext_vector_type(8)))  _Float16 v8h;
typedef __attribute__((ext_vector_type(8)))  float    v8f;

#define C_DIM 256
#define N_TOK 4096
#define B_DIM 4

// padded LDS row strides (halves) to avoid 64-bank conflicts
#define KT_STRIDE 264   // 256 + 8
#define VT_STRIDE 40    // 32 + 8

__device__ inline v8f wmma_f16(v16h a, v16h b, v8f c) {
  // (neg_a, A, neg_b, B, c_mod, C, reuse_a, reuse_b)
  return __builtin_amdgcn_wmma_f32_16x16x32_f16(false, a, false, b, (short)0, c,
                                                false, false);
}

// 16-half load/store in two 16B pieces (safe for 16B-aligned padded strides)
__device__ inline v16h ld16h(const _Float16* p) {
  v8h a = *(const v8h*)p;
  v8h b = *(const v8h*)(p + 8);
  v16h r;
#pragma unroll
  for (int i = 0; i < 8; ++i) { r[i] = a[i]; r[i + 8] = b[i]; }
  return r;
}
__device__ inline void st16h(_Float16* p, v16h v) {
  v8h a, b;
#pragma unroll
  for (int i = 0; i < 8; ++i) { a[i] = v[i]; b[i] = v[i + 8]; }
  *(v8h*)p = a;
  *(v8h*)(p + 8) = b;
}

// A-fragment (16x32 f16, M=rows, K=contraction) from an f16 row (stride s halves).
// Lane l holds row M=(l&15); halves 0..15 -> K=16*hi+0..15
__device__ inline v16h load_a_f16(const _Float16* __restrict__ row, int kbase, int hi) {
  v8h a0 = *(const v8h*)(row + kbase + 16 * hi);
  v8h a1 = *(const v8h*)(row + kbase + 16 * hi + 8);
  v16h r;
#pragma unroll
  for (int i = 0; i < 8; ++i) { r[i] = a0[i]; r[i + 8] = a1[i]; }
  return r;
}

// Same A-fragment but converting from an f32 row.
__device__ inline v16h load_a_f32(const float* __restrict__ row, int kbase, int hi) {
  const float* p0 = row + kbase + 16 * hi;
  const float* p1 = p0 + 8;
  v16h r;
#pragma unroll
  for (int i = 0; i < 8; ++i) {
    r[i]     = (_Float16)p0[i];
    r[i + 8] = (_Float16)p1[i];
  }
  return r;
}

// reductions over the 16-lane half-groups (wave32)
__device__ inline float rowmax16(float v) {
#pragma unroll
  for (int m = 1; m < 16; m <<= 1) v = fmaxf(v, __shfl_xor(v, m, 32));
  return v;
}
__device__ inline float rowsum16(float v) {
#pragma unroll
  for (int m = 1; m < 16; m <<= 1) v += __shfl_xor(v, m, 32);
  return v;
}

// mode 0: token == qi (global attention). mode 1: invert patchify.
__device__ inline int map_token(int mode, int bu, int qi, int* b_out) {
  if (mode == 0) { *b_out = bu; return qi; }
  int b = bu >> 2, p = bu & 3;
  int ir = qi >> 5, ic = qi & 31;
  int row = ((p >> 1) << 5) + ir;
  int col = ((p & 1) << 5) + ic;
  *b_out = b;
  return (row << 6) + col;
}

// ---------------------------------------------------------------------------
// Weight prep: W[in][out] f32 -> Wt[out][in] f16
__global__ void mssa_wprep(const float* __restrict__ W, _Float16* __restrict__ Wt) {
  int idx = blockIdx.x * blockDim.x + threadIdx.x;  // 65536
  int o = idx >> 8, i = idx & 255;
  Wt[o * 256 + i] = (_Float16)W[i * 256 + o];
}

// Gate: s[t] = gelu_exact(X[t,:] . ws + bs), one wave per token (coalesced)
__global__ void mssa_gate(const float* __restrict__ X, const float* __restrict__ ws,
                          const float* __restrict__ bs, float* __restrict__ s) {
  int wave = threadIdx.x >> 5, lane = threadIdx.x & 31;
  int t = blockIdx.x * (blockDim.x >> 5) + wave;
  const float* row = X + (size_t)t * C_DIM;
  float acc = 0.f;
#pragma unroll
  for (int j = 0; j < 8; ++j) acc += row[lane + 32 * j] * ws[lane + 32 * j];
#pragma unroll
  for (int m = 1; m < 32; m <<= 1) acc += __shfl_xor(acc, m, 32);
  acc += bs[0];
  if (lane == 0) s[t] = 0.5f * acc * (1.0f + erff(acc * 0.70710678118654752f));
}

// ---------------------------------------------------------------------------
// Projection: Out = A @ W + b, one wave per 16x16 tile, K=256 via 8 WMMAs.
//   X   : f32 [B][4096][256], rows selected through map_token
//   Wt  : f16 [256 out][256 in]
//   Out : transposed==0 -> f16 [bu][nq][256]; transposed==1 -> f16 [bu][256][nq]
__global__ void mssa_proj(const float* __restrict__ X, const _Float16* __restrict__ Wt,
                          const float* __restrict__ bias, _Float16* __restrict__ Out,
                          int transposed, int mode, int nq) {
  int lane = threadIdx.x & 31;
  int hi = lane >> 4, ln = lane & 15;
  int ct = blockIdx.x & 15;
  int rest = blockIdx.x >> 4;
  int tilesPerBU = nq >> 4;
  int bu = rest / tilesPerBU, tt = rest % tilesPerBU;

  int b, token;
  token = map_token(mode, bu, tt * 16 + ln, &b);
  const float* arow = X + ((size_t)b * N_TOK + token) * C_DIM;

  v8f acc = {0.f, 0.f, 0.f, 0.f, 0.f, 0.f, 0.f, 0.f};
#pragma unroll
  for (int ck = 0; ck < 8; ++ck) {
    v16h a = load_a_f32(arow, 32 * ck, hi);
    v16h bb = ld16h(Wt + (size_t)(16 * ct + ln) * 256 + 32 * ck + 16 * hi);
    acc = wmma_f16(a, bb, acc);
  }

  float bv = bias[16 * ct + ln];
  _Float16* Ob = Out + (size_t)bu * nq * C_DIM;
  if (!transposed) {
#pragma unroll
    for (int r = 0; r < 8; ++r)
      Ob[(size_t)(tt * 16 + r + 8 * hi) * C_DIM + 16 * ct + ln] = (_Float16)(acc[r] + bv);
  } else {
    v8h pk;
#pragma unroll
    for (int r = 0; r < 8; ++r) pk[r] = (_Float16)(acc[r] + bv);
    *(v8h*)(Ob + (size_t)(16 * ct + ln) * nq + tt * 16 + 8 * hi) = pk;
  }
}

// ---------------------------------------------------------------------------
// Flash attention + gated residual. 4 waves per workgroup (64 queries/WG),
// K/V tiles of 32 keys staged cooperatively into LDS once per WG.
//   Q  : f16 [bu][nq][256]       K : f16 [bu][nkv][256]
//   Vt : f16 [bu][256][nkv]
//   gate/resid/out : f32 in global [B][4096][256] token order
//   out[tok] = gate[tok] * softmax(QK^T/16) V + resid[tok]
__global__ void __launch_bounds__(128)
mssa_attn(const _Float16* __restrict__ Q, const _Float16* __restrict__ K,
          const _Float16* __restrict__ Vt, const float* __restrict__ gate,
          const float* __restrict__ resid, float* __restrict__ outp,
          int nq, int nkv, int mode) {
  int tid = threadIdx.x;
  int wave = tid >> 5;
  int lane = tid & 31;
  int hi = lane >> 4, ln = lane & 15;

  int chunksPerBU = nq >> 6;            // 64 queries per WG
  int bu = blockIdx.x / chunksPerBU;
  int qt = (blockIdx.x % chunksPerBU) * 4 + wave;  // 16-query tile of this wave

  const _Float16* Qb  = Q  + (size_t)bu * nq * C_DIM;
  const _Float16* Kb  = K  + (size_t)bu * nkv * C_DIM;
  const _Float16* Vtb = Vt + (size_t)bu * C_DIM * nkv;

  __shared__ _Float16 ktile[32 * KT_STRIDE];    // [key][ch], padded
  __shared__ _Float16 vtile[256 * VT_STRIDE];   // [ch][key], padded
  __shared__ _Float16 ptile[4 * 16 * 32];       // per-wave P staging
  _Float16* pw = &ptile[wave * 16 * 32];

  // staging indices (128 threads)
  int srow = tid >> 2;            // K tile row (key)    0..31
  int scb  = (tid & 3) * 64;      // K tile channel base
  int sch  = tid * 2;             // V tile channel pair 0..254

  // Q fragments for this 16-row tile (K = 256 -> 8 chunks)
  v16h qf[8];
  const _Float16* qrow = Qb + (size_t)(qt * 16 + ln) * C_DIM;
#pragma unroll
  for (int ck = 0; ck < 8; ++ck) qf[ck] = load_a_f16(qrow, 32 * ck, hi);

  v8f acc[16];
#pragma unroll
  for (int ct = 0; ct < 16; ++ct) acc[ct] = v8f{0.f, 0.f, 0.f, 0.f, 0.f, 0.f, 0.f, 0.f};
  float mrow[8], ssum[8];
#pragma unroll
  for (int r = 0; r < 8; ++r) { mrow[r] = -1e30f; ssum[r] = 0.f; }

  const float nf = 0.0625f;  // 256^-0.5

  for (int kt = 0; kt < nkv; kt += 32) {
    // ---- prefetch next tile into cache (global_prefetch_b8)
    if (kt + 32 < nkv) {
      __builtin_prefetch(Kb + (size_t)(kt + 32 + srow) * C_DIM + scb, 0, 0);
      __builtin_prefetch(Vtb + (size_t)sch * nkv + kt + 32, 0, 0);
    }

    // ---- cooperative staging of K (32x256) and V (256x32) tiles into LDS
    {
      const _Float16* gk = Kb + (size_t)(kt + srow) * C_DIM + scb;
      v16h k0 = ld16h(gk);
      v16h k1 = ld16h(gk + 32);
      const _Float16* gv0 = Vtb + (size_t)sch * nkv + kt;
      const _Float16* gv1 = Vtb + (size_t)(sch + 1) * nkv + kt;
      v16h va = ld16h(gv0), vb = ld16h(gv0 + 16);
      v16h vc = ld16h(gv1), vd = ld16h(gv1 + 16);
      st16h(&ktile[srow * KT_STRIDE + scb], k0);
      st16h(&ktile[srow * KT_STRIDE + scb + 32], k1);
      st16h(&vtile[sch * VT_STRIDE], va);
      st16h(&vtile[sch * VT_STRIDE + 16], vb);
      st16h(&vtile[(sch + 1) * VT_STRIDE], vc);
      st16h(&vtile[(sch + 1) * VT_STRIDE + 16], vd);
    }
    __syncthreads();

    // ---- scores: S[16q x 32k], two 16x16 D tiles, K from LDS
    v8f s0 = {0.f, 0.f, 0.f, 0.f, 0.f, 0.f, 0.f, 0.f};
    v8f s1 = {0.f, 0.f, 0.f, 0.f, 0.f, 0.f, 0.f, 0.f};
#pragma unroll
    for (int ck = 0; ck < 8; ++ck) {
      v16h b0 = ld16h(&ktile[ln * KT_STRIDE + 32 * ck + 16 * hi]);
      v16h b1 = ld16h(&ktile[(ln + 16) * KT_STRIDE + 32 * ck + 16 * hi]);
      s0 = wmma_f16(qf[ck], b0, s0);
      s1 = wmma_f16(qf[ck], b1, s1);
    }

    // ---- online softmax (row = r + 8*hi, cols striped over 16 lanes)
    float p0a[8], p1a[8], corr[8];
#pragma unroll
    for (int r = 0; r < 8; ++r) {
      float a0 = s0[r] * nf, a1 = s1[r] * nf;
      float tmax = rowmax16(fmaxf(a0, a1));
      float nm = fmaxf(mrow[r], tmax);
      corr[r] = __expf(mrow[r] - nm);
      mrow[r] = nm;
      float e0 = __expf(a0 - nm), e1 = __expf(a1 - nm);
      ssum[r] = ssum[r] * corr[r] + rowsum16(e0 + e1);
      p0a[r] = e0; p1a[r] = e1;
    }
#pragma unroll
    for (int ct = 0; ct < 16; ++ct)
#pragma unroll
      for (int r = 0; r < 8; ++r) acc[ct][r] *= corr[r];

    // ---- P: D-layout -> A-layout via per-wave LDS region
#pragma unroll
    for (int r = 0; r < 8; ++r) {
      pw[(r + 8 * hi) * 32 + ln]      = (_Float16)p0a[r];
      pw[(r + 8 * hi) * 32 + 16 + ln] = (_Float16)p1a[r];
    }
    __syncthreads();
    v16h pf = load_a_f16(&pw[ln * 32], 0, hi);

    // ---- O += P @ V (16 output tiles over 256 channels), V from LDS
#pragma unroll
    for (int ct = 0; ct < 16; ++ct) {
      v16h bv = ld16h(&vtile[(16 * ct + ln) * VT_STRIDE + 16 * hi]);
      acc[ct] = wmma_f16(pf, bv, acc[ct]);
    }
    __syncthreads();  // protect tiles before next staging round
  }

  // ---- epilogue: normalize, gate, residual
#pragma unroll
  for (int r = 0; r < 8; ++r) {
    int b, token;
    token = map_token(mode, bu, qt * 16 + r + 8 * hi, &b);
    size_t base = ((size_t)b * N_TOK + token) * C_DIM;
    float g = gate[(size_t)b * N_TOK + token];
    float inv = 1.0f / ssum[r];
#pragma unroll
    for (int ct = 0; ct < 16; ++ct) {
      int c = 16 * ct + ln;
      outp[base + c] = g * (acc[ct][r] * inv) + resid[base + c];
    }
  }
}

// ---------------------------------------------------------------------------
extern "C" void kernel_launch(void* const* d_in, const int* in_sizes, int n_in,
                              void* d_out, int out_size, void* d_ws, size_t ws_size,
                              hipStream_t stream) {
  const float* x   = (const float*)d_in[0];
  const float* y   = (const float*)d_in[1];
  const float* wq1 = (const float*)d_in[2];
  const float* bq1 = (const float*)d_in[3];
  const float* wk1 = (const float*)d_in[4];
  const float* bk1 = (const float*)d_in[5];
  const float* wv1 = (const float*)d_in[6];
  const float* bv1 = (const float*)d_in[7];
  const float* wq2 = (const float*)d_in[8];
  const float* bq2 = (const float*)d_in[9];
  const float* wk2 = (const float*)d_in[10];
  const float* bk2 = (const float*)d_in[11];
  const float* wv2 = (const float*)d_in[12];
  const float* bv2 = (const float*)d_in[13];
  const float* ws1 = (const float*)d_in[14];
  const float* bs1 = (const float*)d_in[15];
  const float* ws2 = (const float*)d_in[16];
  const float* bs2 = (const float*)d_in[17];
  float* out = (float*)d_out;

  // ---- workspace layout
  const size_t WSZ  = 256 * 256;                      // one weight matrix (halves)
  const size_t PBUF = (size_t)B_DIM * N_TOK * C_DIM;  // one projection buffer (halves)
  _Float16* wt  = (_Float16*)d_ws;                    // 6 * WSZ halves
  _Float16* q1  = wt + 6 * WSZ;
  _Float16* k1  = q1 + PBUF;
  _Float16* v1t = k1 + PBUF;
  _Float16* q2  = v1t + PBUF;
  _Float16* k2  = q2 + PBUF;
  _Float16* v2t = k2 + PBUF;
  float* fbase = (float*)(v2t + PBUF);
  float* s1 = fbase;                                  // B*N
  float* s2 = s1 + (size_t)B_DIM * N_TOK;
  float* x1 = s2 + (size_t)B_DIM * N_TOK;             // B*N*C f32

  // ---- 1. weight prep (6 transposed f16 copies)
  mssa_wprep<<<256, 256, 0, stream>>>(wq1, wt + 0 * WSZ);
  mssa_wprep<<<256, 256, 0, stream>>>(wk1, wt + 1 * WSZ);
  mssa_wprep<<<256, 256, 0, stream>>>(wv1, wt + 2 * WSZ);
  mssa_wprep<<<256, 256, 0, stream>>>(wq2, wt + 3 * WSZ);
  mssa_wprep<<<256, 256, 0, stream>>>(wk2, wt + 4 * WSZ);
  mssa_wprep<<<256, 256, 0, stream>>>(wv2, wt + 5 * WSZ);

  // ---- 2. gate s1 = gelu(x @ ws1 + bs1)   (8 waves/block, 1 wave/token)
  mssa_gate<<<(B_DIM * N_TOK) / 8, 256, 0, stream>>>(x, ws1, bs1, s1);

  // ---- 3. stage-1 projections (4 batch units, nq=4096)
  int pj1_blocks = B_DIM * (N_TOK / 16) * 16;
  mssa_proj<<<pj1_blocks, 32, 0, stream>>>(y, wt + 0 * WSZ, bq1, q1,  0, 0, N_TOK);
  mssa_proj<<<pj1_blocks, 32, 0, stream>>>(x, wt + 1 * WSZ, bk1, k1,  0, 0, N_TOK);
  mssa_proj<<<pj1_blocks, 32, 0, stream>>>(x, wt + 2 * WSZ, bv1, v1t, 1, 0, N_TOK);

  // ---- 4. global flash attention -> x1 = s1 * attn(q1,k1,v1) + x
  mssa_attn<<<B_DIM * (N_TOK / 64), 128, 0, stream>>>(q1, k1, v1t, s1, x, x1,
                                                      N_TOK, N_TOK, 0);

  // ---- 5. gate s2 = gelu(x1 @ ws2 + bs2)
  mssa_gate<<<(B_DIM * N_TOK) / 8, 256, 0, stream>>>(x1, ws2, bs2, s2);

  // ---- 6. stage-2 projections (16 batch units = 4 batches x 4 patches, nq=1024)
  int nq2 = 1024;
  int pj2_blocks = (B_DIM * 4) * (nq2 / 16) * 16;
  mssa_proj<<<pj2_blocks, 32, 0, stream>>>(y,  wt + 3 * WSZ, bq2, q2,  0, 1, nq2);
  mssa_proj<<<pj2_blocks, 32, 0, stream>>>(x1, wt + 4 * WSZ, bk2, k2,  0, 1, nq2);
  mssa_proj<<<pj2_blocks, 32, 0, stream>>>(x1, wt + 5 * WSZ, bv2, v2t, 1, 1, nq2);

  // ---- 7. patch-local flash attention -> out = s2 * attn(q2,k2,v2) + x1
  mssa_attn<<<(B_DIM * 4) * (nq2 / 64), 128, 0, stream>>>(q2, k2, v2t, s2, x1, out,
                                                          nq2, nq2, 1);
  (void)in_sizes; (void)n_in; (void)out_size; (void)ws_size;
}
